// TripletLossX_40664750359407
// MI455X (gfx1250) — compile-verified, matrix operand order checked by reading, not run on hardware
//
#include <hip/hip_runtime.h>
#include <hip/hip_bf16.h>

typedef __attribute__((ext_vector_type(16))) __bf16 v16bf;
typedef __attribute__((ext_vector_type(8)))  float  v8f;

using u32 = unsigned int;

#define NPTS 8192
#define DIM  2048
#define DIM2 (DIM / 2)          // packed uint (2 bf16) per row
#define KSTEPS (DIM / 32)       // 64 k-chunks of 32
#define MBLK 64                 // rows per workgroup
#define NBLK 128                // cols per chunk
#define NSEG 4                  // column segments (grid.y)
#define CPS  ((NPTS / NBLK) / NSEG)   // chunks per segment = 16
#define INF_F __builtin_inff()

union Frag { v16bf v; u32 u[8]; };

__device__ __forceinline__ u32 f32_to_bf16_rne(float f) {
    u32 u = __float_as_uint(f);
    u32 r = u + 0x7fffu + ((u >> 16) & 1u);
    return r >> 16;
}
__device__ __forceinline__ float bf16bits_to_f32(u32 b) {
    return __uint_as_float(b << 16);
}

// CDNA5 async global->LDS copy (ASYNCcnt-tracked, no VGPR round trip).
// vdst = 32-bit LDS byte address, saddr = 64-bit base, vaddr = 32-bit offset.
__device__ __forceinline__ void async_b128(u32 lds_off, const u32* base, u32 byte_off) {
    asm volatile("global_load_async_to_lds_b128 %0, %1, %2"
                 :
                 : "v"(lds_off), "v"(byte_off), "s"(base)
                 : "memory");
}

// ---------------------------------------------------------------------------
// Kernel 1: L2-normalize each row, emit packed bf16 hi/lo panels + sq per row
// ---------------------------------------------------------------------------
__global__ __launch_bounds__(256)
void row_norm_pack(const float* __restrict__ feats,
                   u32* __restrict__ xhi, u32* __restrict__ xlo,
                   float* __restrict__ sq) {
    const int row = blockIdx.x;
    const int t   = threadIdx.x;
    const float* f = feats + (size_t)row * DIM + t * 8;

    float v[8];
    float s = 0.f;
#pragma unroll
    for (int i = 0; i < 8; ++i) { v[i] = f[i]; s += v[i] * v[i]; }

    __shared__ float red[256];
    red[t] = s;
    __syncthreads();
    for (int o = 128; o > 0; o >>= 1) {
        if (t < o) red[t] += red[t + o];
        __syncthreads();
    }
    const float ss  = red[0];
    const float nrm = sqrtf(ss);
    const float inv = 1.0f / fmaxf(nrm, 1e-12f);
    if (t == 0) sq[row] = ss * inv * inv;   // == sum(x*x) of normalized row

    u32* ph = xhi + (size_t)row * DIM2 + t * 4;
    u32* pl = xlo + (size_t)row * DIM2 + t * 4;
#pragma unroll
    for (int p = 0; p < 4; ++p) {
        u32 oh = 0, ol = 0;
#pragma unroll
        for (int e = 0; e < 2; ++e) {
            float x   = v[2 * p + e] * inv;
            u32 hb    = f32_to_bf16_rne(x);
            float xr  = x - bf16bits_to_f32(hb);
            u32 lb    = f32_to_bf16_rne(xr);
            oh |= hb << (16 * e);
            ol |= lb << (16 * e);
        }
        ph[p] = oh;
        pl[p] = ol;
    }
}

// ---------------------------------------------------------------------------
// top-5 register maintenance
// ---------------------------------------------------------------------------
__device__ __forceinline__ void insert_max5(float (&a)[5], float d) {
    float m = a[0]; int mi = 0;
#pragma unroll
    for (int q = 1; q < 5; ++q) if (a[q] < m) { m = a[q]; mi = q; }
    if (d > m) a[mi] = d;
}
__device__ __forceinline__ void insert_min5(float (&a)[5], float d) {
    float m = a[0]; int mi = 0;
#pragma unroll
    for (int q = 1; q < 5; ++q) if (a[q] > m) { m = a[q]; mi = q; }
    if (d < m) a[mi] = d;
}

// ---------------------------------------------------------------------------
// Kernel 2: fused Gram GEMM (bf16 hi/lo x3 WMMA, async double-buffered LDS)
//           + masked partial top-5 per column segment
// ---------------------------------------------------------------------------
__global__ __launch_bounds__(256)
void gram_topk(const u32* __restrict__ xhi, const u32* __restrict__ xlo,
               const float* __restrict__ sq,
               const int* __restrict__ labels, const int* __restrict__ camids,
               float* __restrict__ pp, float* __restrict__ pn) {
    const int tid  = threadIdx.x;
    const int mb   = blockIdx.x;          // 64-row block
    const int seg  = blockIdx.y;          // column segment
    const int wave = tid >> 5;
    const int lane = tid & 31;
    const int msub  = wave & 3;           // M subtile 0..3
    const int nsub0 = (wave >> 2) * 4;    // first of 4 N subtiles

    __shared__ u32 Ah[2][MBLK][16], Al[2][MBLK][16];
    __shared__ u32 Bh[2][NBLK][16], Bl[2][NBLK][16];
    __shared__ float Cd[MBLK][NBLK + 4];
    __shared__ int   jlab[NBLK], jcam[NBLK];
    __shared__ float sqj[NBLK];

    // per-thread staging coordinates
    const int lar = tid >> 2, lac = (tid & 3) * 4;   // A: 64 rows x 16 uints
    const int lbr = tid >> 1, lbc = (tid & 1) * 8;   // B: 128 rows x 16 uints

    auto issue = [&](int nb, int ks, int buf) {
        u32 gA = (u32)(((u32)(mb * MBLK + lar) * DIM2 + ks * 16 + lac) * 4u);
        async_b128((u32)(uintptr_t)&Ah[buf][lar][lac], xhi, gA);
        async_b128((u32)(uintptr_t)&Al[buf][lar][lac], xlo, gA);
        u32 gB = (u32)(((u32)(nb * NBLK + lbr) * DIM2 + ks * 16 + lbc) * 4u);
        async_b128((u32)(uintptr_t)&Bh[buf][lbr][lbc],     xhi, gB);
        async_b128((u32)(uintptr_t)&Bh[buf][lbr][lbc + 4], xhi, gB + 16u);
        async_b128((u32)(uintptr_t)&Bl[buf][lbr][lbc],     xlo, gB);
        async_b128((u32)(uintptr_t)&Bl[buf][lbr][lbc + 4], xlo, gB + 16u);
    };

    // per-row top-k state (threads 0..63 own one row each)
    float tp[5], tn[5];
    int gi = 0, ilab = 0, icam = 0;
    float sqi = 0.f;
    if (tid < MBLK) {
        gi = mb * MBLK + tid;
        ilab = labels[gi]; icam = camids[gi]; sqi = sq[gi];
#pragma unroll
        for (int q = 0; q < 5; ++q) { tp[q] = -INF_F; tn[q] = INF_F; }
    }

    const int ar = msub * 16 + (lane & 15);
    const int b0 = (lane >> 4) * 4;

    for (int nb = seg * CPS; nb < (seg + 1) * CPS; ++nb) {
        if (tid < NBLK) {
            int gj = nb * NBLK + tid;
            jlab[tid] = labels[gj]; jcam[tid] = camids[gj]; sqj[tid] = sq[gj];
        }
        v8f zero = {};
        v8f acc[4];
#pragma unroll
        for (int j = 0; j < 4; ++j) acc[j] = zero;

        issue(nb, 0, 0);                       // prologue: fill buffer 0

        for (int ks = 0; ks < KSTEPS; ++ks) {
            const int cur = ks & 1;
            if (ks + 1 < KSTEPS) {
                issue(nb, ks + 1, cur ^ 1);    // prefetch next K-chunk
                asm volatile("s_wait_asynccnt 0x6" ::: "memory");
            } else {
                asm volatile("s_wait_asynccnt 0x0" ::: "memory");
            }
            __syncthreads();                   // tiles for `cur` visible to all

            Frag ah, al;
#pragma unroll
            for (int i = 0; i < 4; ++i) {
                ah.u[i]     = Ah[cur][ar][b0 + i];
                ah.u[4 + i] = Ah[cur][ar][b0 + 8 + i];
                al.u[i]     = Al[cur][ar][b0 + i];
                al.u[4 + i] = Al[cur][ar][b0 + 8 + i];
            }
#pragma unroll
            for (int j = 0; j < 4; ++j) {
                const int br = (nsub0 + j) * 16 + (lane & 15);
                Frag bh, bl;
#pragma unroll
                for (int i = 0; i < 4; ++i) {
                    bh.u[i]     = Bh[cur][br][b0 + i];
                    bh.u[4 + i] = Bh[cur][br][b0 + 8 + i];
                    bl.u[i]     = Bl[cur][br][b0 + i];
                    bl.u[4 + i] = Bl[cur][br][b0 + 8 + i];
                }
                // dot = hi*hi + hi*lo + lo*hi  (bf16x3 fp32 recovery)
                acc[j] = __builtin_amdgcn_wmma_f32_16x16x32_bf16(
                    false, ah.v, false, bh.v, (short)0, acc[j], false, false);
                acc[j] = __builtin_amdgcn_wmma_f32_16x16x32_bf16(
                    false, ah.v, false, bl.v, (short)0, acc[j], false, false);
                acc[j] = __builtin_amdgcn_wmma_f32_16x16x32_bf16(
                    false, al.v, false, bh.v, (short)0, acc[j], false, false);
            }
            __syncthreads();                   // all reads of `cur` done
        }

        // write dot tile to LDS
        {
            const int n  = lane & 15;
            const int m8 = (lane >> 4) * 8;
#pragma unroll
            for (int j = 0; j < 4; ++j)
#pragma unroll
                for (int r = 0; r < 8; ++r)
                    Cd[msub * 16 + m8 + r][(nsub0 + j) * 16 + n] = acc[j][r];
        }
        __syncthreads();

        // per-row masked top-5 update
        if (tid < MBLK) {
            for (int c = 0; c < NBLK; ++c) {
                int gj = nb * NBLK + c;
                if (gj == gi) continue;                       // ~eye
                float dot = Cd[tid][c];
                float d2  = sqi + sqj[c] - 2.0f * dot;
                float d   = sqrtf(fmaxf(d2, 1e-12f));
                bool sl = (jlab[c] == ilab);
                bool sc = (jcam[c] == icam);
                if (sl) {
                    if (!sc) insert_max5(tp, d);              // cross-cam positives
                } else {
                    float v = sc ? d * (1.0f / 1.2f) : d;     // same-cam neg boost
                    insert_min5(tn, v);
                }
            }
        }
        __syncthreads();   // protect Cd/jlab before next chunk overwrites them
    }

    // write partial top-5 sets for this segment
    if (tid < MBLK) {
        float* op = pp + ((size_t)seg * NPTS + gi) * 5;
        float* on = pn + ((size_t)seg * NPTS + gi) * 5;
#pragma unroll
        for (int q = 0; q < 5; ++q) { op[q] = tp[q]; on[q] = tn[q]; }
    }
}

// ---------------------------------------------------------------------------
// Kernel 3: merge per-segment partial top-5 sets -> d_ap / d_an
// ---------------------------------------------------------------------------
__global__ __launch_bounds__(256)
void merge_topk(const float* __restrict__ pp, const float* __restrict__ pn,
                float* __restrict__ dap_out, float* __restrict__ dan_out) {
    const int r = blockIdx.x * 256 + threadIdx.x;
    float tp[5], tn[5];
#pragma unroll
    for (int q = 0; q < 5; ++q) { tp[q] = -INF_F; tn[q] = INF_F; }
#pragma unroll
    for (int s = 0; s < NSEG; ++s) {
        const float* ip = pp + ((size_t)s * NPTS + r) * 5;
        const float* in_ = pn + ((size_t)s * NPTS + r) * 5;
#pragma unroll
        for (int q = 0; q < 5; ++q) {
            insert_max5(tp, ip[q]);
            insert_min5(tn, in_[q]);
        }
    }
    float sp = 0.f; int cp = 0;
    float sn = 0.f; int cn = 0;
#pragma unroll
    for (int q = 0; q < 5; ++q) {
        if (tp[q] > -INF_F) { sp += tp[q]; ++cp; }
        if (tn[q] <  INF_F) { sn += tn[q]; ++cn; }
    }
    dap_out[r] = (cp > 0) ? sp / (float)cp : -INF_F;
    dan_out[r] = (cn > 0) ? sn / (float)cn :  INF_F;
}

// ---------------------------------------------------------------------------
// Kernel 4: hardness weighting + margin/softplus loss reduction
// ---------------------------------------------------------------------------
__global__ __launch_bounds__(256)
void finalize(const float* __restrict__ dap, const float* __restrict__ dan,
              const int* __restrict__ epoch, float* __restrict__ out) {
    const int t  = threadIdx.x;
    const int ep = *epoch;
    float sw = 0.f, swp = 0.f;
    for (int i = t; i < NPTS; i += 256) {
        float diff = dap[i] - dan[i];
        float w = 1.0f / (1.0f + __expf(-2.0f * diff));       // sigmoid(ALPHA*diff)
        float per;
        if (ep < 10) {
            per = (diff > 0.f) ? diff + log1pf(__expf(-diff))
                               : log1pf(__expf(diff));        // softplus
        } else {
            per = fmaxf(diff + 0.3f, 0.f);                    // margin ranking
        }
        sw  += w;
        swp += w * per;
    }
    __shared__ float r1[256], r2[256];
    r1[t] = sw; r2[t] = swp;
    __syncthreads();
    for (int o = 128; o > 0; o >>= 1) {
        if (t < o) { r1[t] += r1[t + o]; r2[t] += r2[t + o]; }
        __syncthreads();
    }
    if (t == 0) {
        float wm = r1[0] / (float)NPTS;
        out[0] = r2[0] / (fmaxf(wm, 1e-12f) * (float)NPTS);
    }
}

// ---------------------------------------------------------------------------
extern "C" void kernel_launch(void* const* d_in, const int* in_sizes, int n_in,
                              void* d_out, int out_size, void* d_ws, size_t ws_size,
                              hipStream_t stream) {
    const float* feats  = (const float*)d_in[0];
    const int*   labels = (const int*)d_in[1];
    const int*   camids = (const int*)d_in[2];
    const int*   epoch  = (const int*)d_in[3];
    float* out = (float*)d_out;   // [0]=loss, [1..N]=d_ap, [1+N..]=d_an

    u32*   xhi = (u32*)d_ws;
    u32*   xlo = xhi + (size_t)NPTS * DIM2;
    float* sq  = (float*)(xlo + (size_t)NPTS * DIM2);
    float* pp  = sq + NPTS;                      // NSEG * NPTS * 5
    float* pn  = pp + (size_t)NSEG * NPTS * 5;

    row_norm_pack<<<NPTS, 256, 0, stream>>>(feats, xhi, xlo, sq);
    gram_topk<<<dim3(NPTS / MBLK, NSEG), 256, 0, stream>>>(
        xhi, xlo, sq, labels, camids, pp, pn);
    merge_topk<<<NPTS / 256, 256, 0, stream>>>(pp, pn, out + 1, out + 1 + NPTS);
    finalize<<<1, 256, 0, stream>>>(out + 1, out + 1 + NPTS, epoch, out);
}
